// VoxelNet_41781441855735
// MI455X (gfx1250) — matrix operand order repaired and checked by reading, not compiled, not run on hardware
//
#include <hip/hip_runtime.h>
#include <hip/hip_bf16.h>

typedef unsigned short u16;
typedef __attribute__((ext_vector_type(16))) __bf16 v16bf;
typedef __attribute__((ext_vector_type(8)))  float  v8f;

#define BN_EPS 1e-5f
#define NPTS   20000LL     // 2 * MAX_PTS
#define NROWS  200000LL    // NPTS * 10

__device__ __forceinline__ u16 f2bf_bits(float f){
    union { float f; unsigned u; } c; c.f = f;
    unsigned u = c.u;
    unsigned r = (u + 0x7FFFu + ((u >> 16) & 1u)) >> 16;
    return (u16)r;
}
__device__ __forceinline__ __bf16 bfbits(u16 b){ return __builtin_bit_cast(__bf16, b); }

// ---------------- zero (128-bit granules) ----------------
__global__ void k_zero_b128(uint4* p, long long n16){
    long long i = (long long)blockIdx.x * blockDim.x + threadIdx.x;
    long long s = (long long)gridDim.x * blockDim.x;
    uint4 z; z.x = z.y = z.z = z.w = 0u;
    for (; i < n16; i += s) p[i] = z;
}

// ---------------- weight prep: linear -> WMMA B-fragment order ----------------
// Wf index = (((kc*nct + ct)*32 + lane)*16 + e); value = W[n*K + k],
// n = ct*16 + (lane&15), k = kc*32 + 8*(lane>>4) + (e&7) + ((e&8)<<1)
__global__ void k_prep_lin_frag(const float* __restrict__ W, u16* __restrict__ Wf,
                                int K, int Nc, long long total){
    long long i = (long long)blockIdx.x * blockDim.x + threadIdx.x;
    if (i >= total) return;
    int nct = Nc >> 4;
    int e = (int)(i & 15);
    int lane = (int)((i >> 4) & 31);
    long long rest = i >> 9;
    int ct = (int)(rest % nct);
    int kc = (int)(rest / nct);
    int hi = lane >> 4;
    int n = ct * 16 + (lane & 15);
    int k = kc * 32 + 8 * hi + (e & 7) + ((e & 8) << 1);
    Wf[i] = f2bf_bits(W[(long long)n * K + k]);
}

// conv weights [O=64, Ci, 3,3,3] -> B-fragment order per (tap, kc):
// Wf index = ((((t*nkc + kc)*4 + ct)*32 + lane)*16 + e)
__global__ void k_prep_conv_frag(const float* __restrict__ W, u16* __restrict__ Wf,
                                 int Ci, long long total){
    long long i = (long long)blockIdx.x * blockDim.x + threadIdx.x;
    if (i >= total) return;
    int nkc = Ci >> 5;
    int e = (int)(i & 15);
    int lane = (int)((i >> 4) & 31);
    long long rest = i >> 9;
    int ct = (int)(rest & 3); rest >>= 2;
    int kc = (int)(rest % nkc);
    int t  = (int)(rest / nkc);
    int hi = lane >> 4;
    int co = ct * 16 + (lane & 15);
    int ci = kc * 32 + 8 * hi + (e & 7) + ((e & 8) << 1);
    Wf[i] = f2bf_bits(W[((long long)co * Ci + ci) * 27 + t]);
}

// ---------------- VFE-1: x[200000,7] @ W[16,7]^T + b ; mask ; BN stats ----------------
__global__ void k_vfe1(const float* __restrict__ lidar, const float* __restrict__ W,
                       const float* __restrict__ bias, float* __restrict__ H,
                       float* __restrict__ maskf, float* __restrict__ stats){
    __shared__ float ss[32];
    if (threadIdx.x < 32) ss[threadIdx.x] = 0.f;
    __syncthreads();
    long long row = (long long)blockIdx.x * blockDim.x + threadIdx.x;
    if (row < NROWS){
        float x[7]; float mx = -1e30f;
        const float* xr = lidar + row * 7;
        #pragma unroll
        for (int i = 0; i < 7; i++){ x[i] = xr[i]; mx = fmaxf(mx, x[i]); }
        maskf[row] = (mx != 0.f) ? 1.f : 0.f;
        #pragma unroll
        for (int o = 0; o < 16; o++){
            float acc = bias[o];
            #pragma unroll
            for (int i = 0; i < 7; i++) acc += x[i] * W[o * 7 + i];
            H[row * 16 + o] = acc;
            atomicAdd(&ss[o], acc);
            atomicAdd(&ss[16 + o], acc * acc);
        }
    }
    __syncthreads();
    if (threadIdx.x < 32) atomicAdd(&stats[threadIdx.x], ss[threadIdx.x]);
}

// ---------------- finalize BN stats: slot[0..C)=sum, [C..2C)=sumsq -> [512..)=mean,rstd ----------------
__global__ void k_finalize(float* slot, int C, float cnt){
    int c = threadIdx.x;
    if (c < C){
        float m = slot[c] / cnt;
        float v = slot[C + c] / cnt - m * m;
        slot[512 + c] = m;
        slot[512 + C + c] = rsqrtf(fmaxf(v, 0.f) + BN_EPS);
    }
}

// ---------------- VFE BN+ReLU + max-over-T concat + mask -> bf16 [p,10,2U] ----------------
__global__ void k_vfe_bn_concat(const float* __restrict__ H, const float* __restrict__ slot,
                                const float* __restrict__ g, const float* __restrict__ bt,
                                const float* __restrict__ maskf, u16* __restrict__ X, int U){
    int p = blockIdx.x; int c = threadIdx.x;  // blockDim == U
    const float* mrs = slot + 512;
    float m = mrs[c], rs = mrs[U + c], gg = g[c], bb = bt[c];
    float v[10]; float mv = 0.f;
    #pragma unroll
    for (int t = 0; t < 10; t++){
        float h = H[((long long)p * 10 + t) * U + c];
        float y = fmaxf((h - m) * rs * gg + bb, 0.f);
        v[t] = y; mv = fmaxf(mv, y);
    }
    #pragma unroll
    for (int t = 0; t < 10; t++){
        float msk = maskf[(long long)p * 10 + t];
        long long o = ((long long)p * 10 + t) * (2 * U);
        X[o + c]     = f2bf_bits(v[t] * msk);
        X[o + U + c] = f2bf_bits(mv * msk);
    }
}

// ---------------- WMMA GEMM (compile-time tile shape): H[M,Nc] = A_bf16[M,K] @ Wf + bias ----------------
// 4 waves/block, one 16xNc tile per wave. NKC/NCT are constants -> acc stays in VGPRs.
template<int K, int Nc>
__global__ __launch_bounds__(128) void k_gemm_wmma(const u16* __restrict__ A,
                            const u16* __restrict__ Wf,
                            const float* __restrict__ bias, float* __restrict__ H,
                            float* __restrict__ stats, long long M){
    constexpr int NKC = K / 32;
    constexpr int NCT = Nc / 16;
    __shared__ float ssum[2 * Nc];
    int tid = threadIdx.x;
    for (int i = tid; i < 2 * Nc; i += blockDim.x) ssum[i] = 0.f;
    __syncthreads();
    int lane = tid & 31, wv = tid >> 5;
    long long tile = (long long)blockIdx.x * 4 + wv;
    int mrow = lane & 15, hi = lane >> 4;
    v8f acc[NCT] = {};
    const u16* arow = A + (tile * 16 + mrow) * K + 8 * hi;
    #pragma unroll
    for (int kc = 0; kc < NKC; kc++){
        v16bf a;
        const u16* ap = arow + kc * 32;
        #pragma unroll
        for (int e = 0; e < 8; e++){ a[e] = bfbits(ap[e]); a[8 + e] = bfbits(ap[16 + e]); }
        #pragma unroll
        for (int ct = 0; ct < NCT; ct++){
            v16bf b;
            const u16* bp = Wf + (((long long)(kc * NCT + ct) * 32 + lane) << 4);
            #pragma unroll
            for (int e = 0; e < 16; e++) b[e] = bfbits(bp[e]);
            acc[ct] = __builtin_amdgcn_wmma_f32_16x16x32_bf16(false, a, false, b,
                                                              (short)0, acc[ct], false, false);
        }
    }
    #pragma unroll
    for (int ct = 0; ct < NCT; ct++){
        int col = ct * 16 + (lane & 15);
        float bs = bias[col];
        float ps = 0.f, pq = 0.f;
        #pragma unroll
        for (int r = 0; r < 8; r++){
            long long row = tile * 16 + r + 8 * hi;
            float v = acc[ct][r] + bs;
            H[row * Nc + col] = v;
            ps += v; pq += v * v;
        }
        atomicAdd(&ssum[col], ps);
        atomicAdd(&ssum[Nc + col], pq);
    }
    __syncthreads();
    for (int i = tid; i < 2 * Nc; i += blockDim.x) atomicAdd(&stats[i], ssum[i]);
}

// ---------------- stage-3: BN+ReLU then max over T -> vwfs[p,128] f32 ----------------
__global__ void k_bn_max(const float* __restrict__ H, const float* __restrict__ slot,
                         const float* __restrict__ g, const float* __restrict__ bt,
                         float* __restrict__ vwfs){
    int p = blockIdx.x, c = threadIdx.x;  // 128 threads
    const float* mrs = slot + 512;
    float m = mrs[c], rs = mrs[128 + c], gg = g[c], bb = bt[c];
    float mv = 0.f;
    #pragma unroll
    for (int t = 0; t < 10; t++){
        float h = H[((long long)p * 10 + t) * 128 + c];
        mv = fmaxf(mv, fmaxf((h - m) * rs * gg + bb, 0.f));
    }
    vwfs[(long long)p * 128 + c] = mv;
}

// ---------------- scatter vwfs -> dense bf16 [b,d,h,w,c] ----------------
__global__ void k_scatter(const float* __restrict__ vwfs, const int* __restrict__ coords,
                          u16* __restrict__ dense){
    long long i = (long long)blockIdx.x * blockDim.x + threadIdx.x;
    if (i >= NPTS * 128) return;
    int c = (int)(i & 127); long long p = i >> 7;
    int b = (int)(p / 10000);
    int d = coords[p * 3 + 0], h = coords[p * 3 + 1], w = coords[p * 3 + 2];
    long long off = ((((long long)b * 8 + d) * 200 + h) * 200 + w) * 128 + c;
    dense[off] = f2bf_bits(vwfs[i]);
}

// ---------------- implicit-GEMM 3D conv via WMMA; channel-last in/out, Cout=64 ----------------
// CI compile-time -> kc/ct loops fully unrolled, acc[4] in VGPRs across all 27 taps.
template<int CI>
__global__ __launch_bounds__(128) void k_conv3d_wmma(const u16* __restrict__ In,
                              const u16* __restrict__ Wf,
                              const float* __restrict__ bias, float* __restrict__ Out,
                              float* __restrict__ stats,
                              int Di, int Hi, int Wi,
                              int Do, int Ho, int Wo,
                              int sd, int sh, int sw, int pd, int ph, int pw){
    constexpr int NKC = CI / 32;
    __shared__ float ssum[128];
    int tid = threadIdx.x;
    for (int i = tid; i < 128; i += blockDim.x) ssum[i] = 0.f;
    __syncthreads();
    int lane = tid & 31, wv = tid >> 5;
    long long tile = (long long)blockIdx.x * 4 + wv;
    int mrow = lane & 15, hi = lane >> 4;
    long long pos = tile * 16 + mrow;
    int ow = (int)(pos % Wo); long long r0 = pos / Wo;
    int oh = (int)(r0 % Ho); r0 /= Ho;
    int od = (int)(r0 % Do); int b = (int)(r0 / Do);
    int id0 = od * sd - pd, ih0 = oh * sh - ph, iw0 = ow * sw - pw;
    v8f acc[4] = {};
    for (int kd = 0; kd < 3; kd++){
        int id = id0 + kd; bool okd = (id >= 0) && (id < Di);
        for (int kh = 0; kh < 3; kh++){
            int ih = ih0 + kh; bool okh = okd && (ih >= 0) && (ih < Hi);
            for (int kw = 0; kw < 3; kw++){
                int iw = iw0 + kw; bool ok = okh && (iw >= 0) && (iw < Wi);
                int t = (kd * 3 + kh) * 3 + kw;
                const u16* rowp = In + ((((long long)b * Di + id) * Hi + ih) * (long long)Wi + iw) * CI + 8 * hi;
                #pragma unroll
                for (int kc = 0; kc < NKC; kc++){
                    v16bf a;
                    if (ok){
                        const u16* ap = rowp + kc * 32;
                        #pragma unroll
                        for (int e = 0; e < 8; e++){ a[e] = bfbits(ap[e]); a[8 + e] = bfbits(ap[16 + e]); }
                    } else {
                        #pragma unroll
                        for (int e = 0; e < 16; e++) a[e] = bfbits((u16)0);
                    }
                    const u16* bbase = Wf + ((long long)(t * NKC + kc) * 4 * 32 + lane) * 16;
                    #pragma unroll
                    for (int ct = 0; ct < 4; ct++){
                        v16bf bw;
                        const u16* bp = bbase + ct * 512;
                        #pragma unroll
                        for (int e = 0; e < 16; e++) bw[e] = bfbits(bp[e]);
                        acc[ct] = __builtin_amdgcn_wmma_f32_16x16x32_bf16(false, a, false, bw,
                                                                          (short)0, acc[ct], false, false);
                    }
                }
            }
        }
    }
    #pragma unroll
    for (int ct = 0; ct < 4; ct++){
        int col = ct * 16 + (lane & 15);
        float bs = bias[col];
        float ps = 0.f, pq = 0.f;
        #pragma unroll
        for (int r = 0; r < 8; r++){
            long long prow = tile * 16 + r + 8 * hi;
            float v = acc[ct][r] + bs;
            Out[prow * 64 + col] = v;
            ps += v; pq += v * v;
        }
        atomicAdd(&ssum[col], ps);
        atomicAdd(&ssum[64 + col], pq);
    }
    __syncthreads();
    for (int i = tid; i < 128; i += blockDim.x) atomicAdd(&stats[i], ssum[i]);
}

// ---------------- conv BN+ReLU apply: -> bf16 channel-last and/or f32 channel-first ----------------
__global__ void k_bn_apply(const float* __restrict__ Y, const float* __restrict__ slot,
                           const float* __restrict__ g, const float* __restrict__ bt,
                           u16* __restrict__ obf, float* __restrict__ ocf,
                           long long M, int S){
    long long i = (long long)blockIdx.x * blockDim.x + threadIdx.x;
    if (i >= M * 64) return;
    int c = (int)(i & 63); long long pos = i >> 6;
    const float* mrs = slot + 512;
    float v = fmaxf((Y[i] - mrs[c]) * mrs[64 + c] * g[c] + bt[c], 0.f);
    if (obf) obf[i] = f2bf_bits(v);
    if (ocf){
        long long b = pos / S, s = pos % S;
        ocf[(b * 64 + c) * S + s] = v;
    }
}

extern "C" void kernel_launch(void* const* d_in, const int* in_sizes, int n_in,
                              void* d_out, int out_size, void* d_ws, size_t ws_size,
                              hipStream_t stream){
    (void)in_sizes; (void)n_in; (void)out_size; (void)ws_size;
    const float* lidar  = (const float*)d_in[0];
    const int*   coords = (const int*)d_in[1];
    const float* v1W = (const float*)d_in[2],  *v1b = (const float*)d_in[3];
    const float* v1g = (const float*)d_in[4],  *v1bt = (const float*)d_in[5];
    const float* v2W = (const float*)d_in[6],  *v2b = (const float*)d_in[7];
    const float* v2g = (const float*)d_in[8],  *v2bt = (const float*)d_in[9];
    const float* fW  = (const float*)d_in[10], *fb  = (const float*)d_in[11];
    const float* fg  = (const float*)d_in[12], *fbt = (const float*)d_in[13];
    const float* c1W = (const float*)d_in[14], *c1b = (const float*)d_in[15];
    const float* c1g = (const float*)d_in[16], *c1bt = (const float*)d_in[17];
    const float* c2W = (const float*)d_in[18], *c2b = (const float*)d_in[19];
    const float* c2g = (const float*)d_in[20], *c2bt = (const float*)d_in[21];
    const float* c3W = (const float*)d_in[22], *c3b = (const float*)d_in[23];
    const float* c3g = (const float*)d_in[24], *c3bt = (const float*)d_in[25];

    char* ws = (char*)d_ws;
    // ---- workspace layout (bytes) ----
    float* stats = (float*)(ws + 0);                    // 6 slots x 1024 f32 = 24 KB
    u16* wfV2 = (u16*)(ws + (32LL << 10));              //   4 KB
    u16* wfF  = (u16*)(ws + (36LL << 10));              //  32 KB
    u16* wfC1 = (u16*)(ws + (68LL << 10));              // 442 KB
    u16* wfC2 = (u16*)(ws + (512LL << 10));             // 221 KB
    u16* wfC3 = (u16*)(ws + (736LL << 10));             // 221 KB
    float* maskf = (float*)(ws + (960LL << 10));        // 800 KB
    const long long MB = 1LL << 20;
    float* h1   = (float*)(ws + 2 * MB);                // 12.8 MB
    u16*   x2   = (u16*)  (ws + 16 * MB);               // 12.8 MB
    float* h2   = (float*)(ws + 30 * MB);               // 51.2 MB
    u16*   x3   = (u16*)  (ws + 84 * MB);               // 51.2 MB
    float* h3   = (float*)(ws + 136 * MB);              // 102.4 MB
    float* vwfs = h1;                                   // reuse (h1 dead)
    u16*   dense = (u16*)(ws + 30 * MB);                // reuse h2..h3 span: 163.84 MB
    float* y1   = (float*)(ws + 240 * MB);              // 81.92 MB
    u16*   y1bf = (u16*)  (ws + 30 * MB);               // reuse (dense dead): 40.96 MB
    float* y2   = (float*)(ws + 80 * MB);               // 40.96 MB
    u16*   y2bf = (u16*)  (ws + 128 * MB);              // 20.48 MB
    float* y3   = y1;                                   // reuse (y1 dead): 20.48 MB
    float* outf = (float*)d_out;                        // [2,64,1,200,200]

    float* slot1 = stats + 0 * 1024;   // vfe1 (C=16)
    float* slot2 = stats + 1 * 1024;   // vfe2 (C=64)
    float* slot3 = stats + 2 * 1024;   // fcn  (C=128)
    float* slot4 = stats + 3 * 1024;   // conv1
    float* slot5 = stats + 4 * 1024;   // conv2
    float* slot6 = stats + 5 * 1024;   // conv3

    // 0) zero stats and dense grid (ws is poisoned by harness)
    k_zero_b128<<<12, 128, 0, stream>>>((uint4*)stats, 6 * 1024 * 4 / 16);
    k_zero_b128<<<4096, 256, 0, stream>>>((uint4*)dense, (2LL * 8 * 200 * 200 * 128 * 2) / 16);

    // 1) weight prep into WMMA B-fragment layouts
    { long long t = 1LL * 4 * 32 * 16;            k_prep_lin_frag<<<(int)((t + 255) / 256), 256, 0, stream>>>(v2W, wfV2, 32, 64, t); }
    { long long t = 4LL * 8 * 32 * 16;            k_prep_lin_frag<<<(int)((t + 255) / 256), 256, 0, stream>>>(fW, wfF, 128, 128, t); }
    { long long t = 27LL * 4 * 4 * 32 * 16;       k_prep_conv_frag<<<(int)((t + 255) / 256), 256, 0, stream>>>(c1W, wfC1, 128, t); }
    { long long t = 27LL * 2 * 4 * 32 * 16;       k_prep_conv_frag<<<(int)((t + 255) / 256), 256, 0, stream>>>(c2W, wfC2, 64, t); }
    { long long t = 27LL * 2 * 4 * 32 * 16;       k_prep_conv_frag<<<(int)((t + 255) / 256), 256, 0, stream>>>(c3W, wfC3, 64, t); }

    // 2) VFE stage 1 (7->16) + stats; finalize; BN+ReLU+concat -> x2 bf16 [.,32]
    k_vfe1<<<(int)((NROWS + 255) / 256), 256, 0, stream>>>(lidar, v1W, v1b, h1, maskf, slot1);
    k_finalize<<<1, 128, 0, stream>>>(slot1, 16, (float)NROWS);
    k_vfe_bn_concat<<<(int)NPTS, 16, 0, stream>>>(h1, slot1, v1g, v1bt, maskf, x2, 16);

    // 3) VFE stage 2 GEMM (32->64) WMMA + stats; BN+ReLU+concat -> x3 bf16 [.,128]
    k_gemm_wmma<32, 64><<<(int)(NROWS / 64), 128, 0, stream>>>(x2, wfV2, v2b, h2, slot2, NROWS);
    k_finalize<<<1, 128, 0, stream>>>(slot2, 64, (float)NROWS);
    k_vfe_bn_concat<<<(int)NPTS, 64, 0, stream>>>(h2, slot2, v2g, v2bt, maskf, x3, 64);

    // 4) FCN GEMM (128->128) WMMA + stats; BN+ReLU + max over T -> vwfs [N,128]
    k_gemm_wmma<128, 128><<<(int)(NROWS / 64), 128, 0, stream>>>(x3, wfF, fb, h3, slot3, NROWS);
    k_finalize<<<1, 128, 0, stream>>>(slot3, 128, (float)NROWS);
    k_bn_max<<<(int)NPTS, 128, 0, stream>>>(h3, slot3, fg, fbt, vwfs);

    // 5) scatter to dense bf16 grid [2,8,200,200,128]
    k_scatter<<<(int)((NPTS * 128 + 255) / 256), 256, 0, stream>>>(vwfs, coords, dense);

    // 6) conv1: 128->64, stride(2,1,1), pad(1,1,1): [2,8,200,200] -> [2,4,200,200]
    {
        long long M = 2LL * 4 * 200 * 200; // 320000
        k_conv3d_wmma<128><<<(int)(M / 64), 128, 0, stream>>>(dense, wfC1, c1b, y1, slot4,
            8, 200, 200, 4, 200, 200, 2, 1, 1, 1, 1, 1);
        k_finalize<<<1, 128, 0, stream>>>(slot4, 64, (float)M);
        k_bn_apply<<<(int)((M * 64 + 255) / 256), 256, 0, stream>>>(y1, slot4, c1g, c1bt,
            y1bf, (float*)nullptr, M, 4 * 200 * 200);
    }
    // 7) conv2: 64->64, stride(1,1,1), pad(0,1,1): [2,4,200,200] -> [2,2,200,200]
    {
        long long M = 2LL * 2 * 200 * 200; // 160000
        k_conv3d_wmma<64><<<(int)(M / 64), 128, 0, stream>>>(y1bf, wfC2, c2b, y2, slot5,
            4, 200, 200, 2, 200, 200, 1, 1, 1, 0, 1, 1);
        k_finalize<<<1, 128, 0, stream>>>(slot5, 64, (float)M);
        k_bn_apply<<<(int)((M * 64 + 255) / 256), 256, 0, stream>>>(y2, slot5, c2g, c2bt,
            y2bf, (float*)nullptr, M, 2 * 200 * 200);
    }
    // 8) conv3: 64->64, stride(2,1,1), pad(1,1,1): [2,2,200,200] -> [2,1,200,200]; emit d_out NCDHW
    {
        long long M = 2LL * 1 * 200 * 200; // 80000
        k_conv3d_wmma<64><<<(int)(M / 64), 128, 0, stream>>>(y2bf, wfC3, c3b, y3, slot6,
            2, 200, 200, 1, 200, 200, 2, 1, 1, 1, 1, 1);
        k_finalize<<<1, 128, 0, stream>>>(slot6, 64, (float)M);
        k_bn_apply<<<(int)((M * 64 + 255) / 256), 256, 0, stream>>>(y3, slot6, c3g, c3bt,
            (u16*)nullptr, outf, M, 1 * 200 * 200);
    }
}
